// Attention_21758304322201
// MI455X (gfx1250) — compile-verified, matrix operand order checked by reading, not run on hardware
//
#include <hip/hip_runtime.h>
#include <hip/hip_bf16.h>

typedef __bf16 bf16_t;
typedef __attribute__((ext_vector_type(16))) __bf16 v16bf;
typedef __attribute__((ext_vector_type(8)))  float  v8f;

#define NB   32
#define LQ   2048
#define LK   2048
#define DIN  32
#define DKQ  20
#define EPSLN 1e-5f

union Frag { uint4 u[2]; v16bf v; };

// ---------------------------------------------------------------------------
// Preprocessing: projections + layernorms, emit bf16 operands for WMMA.
//   mode 0: q = LN(ques @ W_q^T)  -> q_bf  [B*LQ][32] (20 ch + zero pad)
//   mode 1: k = LN(keys @ W_k^T)  -> k_bf  [B*LK][32] (20 ch + zero pad)
//   mode 2: v = vals @ W_v^T      -> vt_bf [B*32][LK] (channel-major)
// One wave per row; lane c computes output channel c via shuffle-broadcast.
// ---------------------------------------------------------------------------
__global__ __launch_bounds__(256) void preproc_kernel(
    const float* __restrict__ vals,
    const float* __restrict__ keys,
    const float* __restrict__ ques,
    const float* __restrict__ W_v,
    const float* __restrict__ W_k,
    const float* __restrict__ W_q,
    const float* __restrict__ g_k, const float* __restrict__ b_k,
    const float* __restrict__ g_q, const float* __restrict__ b_q,
    bf16_t* __restrict__ q_bf,
    bf16_t* __restrict__ k_bf,
    bf16_t* __restrict__ vt_bf)
{
    const int mode = blockIdx.y;            // 0=q, 1=k, 2=v
    const int wave = threadIdx.x >> 5;
    const int lane = threadIdx.x & 31;
    const int row  = blockIdx.x * 8 + wave; // [0, NB*2048)

    const float* src = (mode == 0) ? ques : (mode == 1) ? keys : vals;
    const float* W   = (mode == 0) ? W_q  : (mode == 1) ? W_k  : W_v;
    const int    od  = (mode == 2) ? DIN : DKQ;

    const float x = src[(size_t)row * DIN + lane];

    float acc = 0.f;
    #pragma unroll
    for (int d = 0; d < DIN; ++d) {
        const float xd = __shfl(x, d, 32);
        const float w  = (lane < od) ? W[lane * DIN + d] : 0.f;
        acc = fmaf(xd, w, acc);
    }

    if (mode == 2) {
        const int bb = row >> 11;
        const int ki = row & (LK - 1);
        vt_bf[((size_t)(bb * DIN + lane)) * LK + ki] = (bf16_t)acc;
        return;
    }

    // LayerNorm over 20 channels (lanes >= 20 contribute zero, store zero pad)
    float s = (lane < DKQ) ? acc : 0.f;
    #pragma unroll
    for (int m = 1; m < 32; m <<= 1) s += __shfl_xor(s, m, 32);
    const float mean = s * (1.f / DKQ);

    float dv = (lane < DKQ) ? (acc - mean) : 0.f;
    float vv = dv * dv;
    #pragma unroll
    for (int m = 1; m < 32; m <<= 1) vv += __shfl_xor(vv, m, 32);
    const float rstd = rsqrtf(vv * (1.f / DKQ) + EPSLN);

    float y = 0.f;
    if (lane < DKQ) {
        const float g  = (mode == 0) ? g_q[lane] : g_k[lane];
        const float bt = (mode == 0) ? b_q[lane] : b_k[lane];
        y = dv * rstd * g + bt;
    }
    bf16_t* dst = (mode == 0) ? q_bf : k_bf;
    dst[(size_t)row * DIN + lane] = (bf16_t)y;
}

// ---------------------------------------------------------------------------
// Flash attention: one wave = 16 query rows; 8 waves/block = 128 rows.
// Streams keys in chunks of 64, online softmax, fp32 accumulators.
// S and O GEMMs both use v_wmma_f32_16x16x32_bf16. P goes through LDS to be
// re-laid-out from C-matrix layout into A-matrix layout.
// Epilogue: O/l + residual + output LayerNorm, all in registers.
// ---------------------------------------------------------------------------
__global__ __launch_bounds__(256) void flash_attn_kernel(
    const bf16_t* __restrict__ q_bf,
    const bf16_t* __restrict__ k_bf,
    const bf16_t* __restrict__ vt_bf,
    const unsigned char* __restrict__ key_mask,
    const float* __restrict__ ques,
    const float* __restrict__ g_o, const float* __restrict__ b_o,
    float* __restrict__ out)
{
    // per-wave P tile: 16 rows x 64 cols bf16, row stride 80 (160B, 16B-aligned)
    __shared__ bf16_t Pbuf[8 * 16 * 80];

    const int  wave  = threadIdx.x >> 5;
    const int  lane  = threadIdx.x & 31;
    const int  ln    = lane & 15;
    const bool hi    = lane >= 16;
    const int  b     = blockIdx.x >> 4;
    const int  qblk  = blockIdx.x & 15;
    const int  qbase = qblk * 128 + wave * 16;

    // ---- load Q A-fragment (16x32 bf16) ----
    // lanes 0-15: row=ln, K chans {0..7,16..23}; lanes 16-31: {8..15,24..31}
    Frag aq;
    {
        const bf16_t* qrow = q_bf + ((size_t)(b * LQ + qbase + ln)) * DIN;
        aq.u[0] = *(const uint4*)(qrow + (hi ? 8 : 0));
        aq.u[1] = *(const uint4*)(qrow + (hi ? 24 : 16));
    }

    v8f acc0 = {};   // output channels 0..15  (C layout)
    v8f acc1 = {};   // output channels 16..31
    float m_i[8], l_i[8];
    #pragma unroll
    for (int r = 0; r < 8; ++r) { m_i[r] = -1e30f; l_i[r] = 0.f; }

    const float scale = 0.22360679774997896f;   // 1/sqrt(20)
    bf16_t* P = Pbuf + wave * (16 * 80);
    const unsigned char* mrow = key_mask + (size_t)b * LK;

    for (int k0 = 0; k0 < LK; k0 += 64) {
        float p[4][8];
        float cmax[8];
        #pragma unroll
        for (int r = 0; r < 8; ++r) cmax[r] = -1e30f;

        // ---- S = q @ K^T over 4 column tiles of 16 keys ----
        #pragma unroll
        for (int ct = 0; ct < 4; ++ct) {
            const int j = k0 + ct * 16 + ln;
            Frag bk;
            const bf16_t* krow = k_bf + ((size_t)(b * LK + j)) * DIN;
            bk.u[0] = *(const uint4*)(krow + (hi ? 8 : 0));
            bk.u[1] = *(const uint4*)(krow + (hi ? 24 : 16));
            v8f cz = {};
            v8f sreg = __builtin_amdgcn_wmma_f32_16x16x32_bf16(
                false, aq.v, false, bk.v, (short)0, cz, false, false);
            const bool msk = mrow[j] != 0;
            #pragma unroll
            for (int r = 0; r < 8; ++r) {
                const float sv = msk ? -1e30f : sreg[r] * scale;
                p[ct][r] = sv;
                cmax[r]  = fmaxf(cmax[r], sv);
            }
        }

        // ---- row max across the 16-lane half (rows 0-7 | rows 8-15) ----
        #pragma unroll
        for (int r = 0; r < 8; ++r) {
            float c = cmax[r];
            c = fmaxf(c, __shfl_xor(c, 1, 32));
            c = fmaxf(c, __shfl_xor(c, 2, 32));
            c = fmaxf(c, __shfl_xor(c, 4, 32));
            c = fmaxf(c, __shfl_xor(c, 8, 32));
            cmax[r] = c;
        }

        // ---- online softmax update ----
        float corr[8];
        #pragma unroll
        for (int r = 0; r < 8; ++r) {
            const float nm = fmaxf(m_i[r], cmax[r]);
            corr[r] = __expf(m_i[r] - nm);
            m_i[r]  = nm;
        }
        #pragma unroll
        for (int ct = 0; ct < 4; ++ct)
            #pragma unroll
            for (int r = 0; r < 8; ++r)
                p[ct][r] = __expf(p[ct][r] - m_i[r]);   // masked -> exp(-1e30-nm)=0

        #pragma unroll
        for (int r = 0; r < 8; ++r) {
            float cs = p[0][r] + p[1][r] + p[2][r] + p[3][r];
            cs += __shfl_xor(cs, 1, 32);
            cs += __shfl_xor(cs, 2, 32);
            cs += __shfl_xor(cs, 4, 32);
            cs += __shfl_xor(cs, 8, 32);
            l_i[r] = l_i[r] * corr[r] + cs;
            acc0[r] *= corr[r];
            acc1[r] *= corr[r];
        }

        // ---- P (C layout) -> LDS row-major [16][80] bf16 ----
        #pragma unroll
        for (int ct = 0; ct < 4; ++ct)
            #pragma unroll
            for (int r = 0; r < 8; ++r)
                P[(r + (hi ? 8 : 0)) * 80 + ct * 16 + ln] = (bf16_t)p[ct][r];

        // ---- O += P @ V : two K-subchunks of 32, two output col tiles ----
        #pragma unroll
        for (int kc = 0; kc < 64; kc += 32) {
            Frag ap;
            const bf16_t* prow = P + ln * 80 + kc + (hi ? 8 : 0);
            ap.u[0] = *(const uint4*)(prow);
            ap.u[1] = *(const uint4*)(prow + 16);

            #pragma unroll
            for (int nt = 0; nt < 2; ++nt) {
                Frag bv;
                const bf16_t* vrow = vt_bf
                    + ((size_t)(b * DIN + nt * 16 + ln)) * LK
                    + k0 + kc + (hi ? 8 : 0);
                bv.u[0] = *(const uint4*)(vrow);
                bv.u[1] = *(const uint4*)(vrow + 16);
                if (nt == 0)
                    acc0 = __builtin_amdgcn_wmma_f32_16x16x32_bf16(
                        false, ap.v, false, bv.v, (short)0, acc0, false, false);
                else
                    acc1 = __builtin_amdgcn_wmma_f32_16x16x32_bf16(
                        false, ap.v, false, bv.v, (short)0, acc1, false, false);
            }
        }
    }

    // ---- epilogue: out = LN(O/l + ques) ----
    const int col0 = ln, col1 = ln + 16;
    const float go0 = g_o[col0], go1 = g_o[col1];
    const float bo0 = b_o[col0], bo1 = b_o[col1];

    #pragma unroll
    for (int r = 0; r < 8; ++r) {
        const int row = qbase + r + (hi ? 8 : 0);
        const float* qq = ques + ((size_t)(b * LQ + row)) * DIN;
        const float invl = 1.0f / l_i[r];
        const float x0 = acc0[r] * invl + qq[col0];
        const float x1 = acc1[r] * invl + qq[col1];

        float s = x0 + x1;
        s += __shfl_xor(s, 1, 32);
        s += __shfl_xor(s, 2, 32);
        s += __shfl_xor(s, 4, 32);
        s += __shfl_xor(s, 8, 32);
        const float mean = s * (1.f / DIN);

        const float d0 = x0 - mean, d1 = x1 - mean;
        float vv = d0 * d0 + d1 * d1;
        vv += __shfl_xor(vv, 1, 32);
        vv += __shfl_xor(vv, 2, 32);
        vv += __shfl_xor(vv, 4, 32);
        vv += __shfl_xor(vv, 8, 32);
        const float rstd = rsqrtf(vv * (1.f / DIN) + EPSLN);

        float* orow = out + ((size_t)(b * LQ + row)) * DIN;
        orow[col0] = d0 * rstd * go0 + bo0;
        orow[col1] = d1 * rstd * go1 + bo1;
    }
}

// ---------------------------------------------------------------------------
extern "C" void kernel_launch(void* const* d_in, const int* in_sizes, int n_in,
                              void* d_out, int out_size, void* d_ws, size_t ws_size,
                              hipStream_t stream) {
    (void)in_sizes; (void)n_in; (void)out_size; (void)ws_size;

    const float* vals = (const float*)d_in[0];
    const float* keys = (const float*)d_in[1];
    const float* ques = (const float*)d_in[2];
    const unsigned char* key_mask = (const unsigned char*)d_in[3];
    const float* W_v  = (const float*)d_in[4];
    const float* W_k  = (const float*)d_in[5];
    const float* W_q  = (const float*)d_in[6];
    const float* g_k  = (const float*)d_in[7];
    const float* b_k  = (const float*)d_in[8];
    const float* g_q  = (const float*)d_in[9];
    const float* b_q  = (const float*)d_in[10];
    const float* g_o  = (const float*)d_in[11];
    const float* b_o  = (const float*)d_in[12];

    bf16_t* q_bf  = (bf16_t*)d_ws;                       // 4 MB
    bf16_t* k_bf  = q_bf  + (size_t)NB * LQ * DIN;       // 4 MB
    bf16_t* vt_bf = k_bf  + (size_t)NB * LK * DIN;       // 4 MB

    dim3 gpre((NB * LK) / 8, 3, 1);
    preproc_kernel<<<gpre, dim3(256), 0, stream>>>(
        vals, keys, ques, W_v, W_k, W_q, g_k, b_k, g_q, b_q,
        q_bf, k_bf, vt_bf);

    dim3 gatt(NB * (LQ / 128), 1, 1);   // 512 blocks, 8 waves each
    flash_attn_kernel<<<gatt, dim3(256), 0, stream>>>(
        q_bf, k_bf, vt_bf, key_mask, ques, g_o, b_o, (float*)d_out);
}
